// MultiHeadSelfAttention_28406913695885
// MI455X (gfx1250) — matrix operand order repaired and checked by reading, not compiled, not run on hardware
//
#include <hip/hip_runtime.h>

#define D_MODEL 1024
#define N_HEADS 16
#define HEAD_DIM 64
#define SEQ 2048
#define BATCH 4
#define NEG_INF -1e30f

typedef __attribute__((ext_vector_type(16))) __bf16 v16bf;
typedef __attribute__((ext_vector_type(8)))  float  v8f;
typedef unsigned short u16;
typedef __attribute__((ext_vector_type(8))) unsigned short us8;
typedef __attribute__((ext_vector_type(4))) unsigned short us4;

__device__ __forceinline__ u16 f2bf(float f) {
  union { float f; unsigned u; } x; x.f = f;
  unsigned r = x.u + 0x7FFFu + ((x.u >> 16) & 1u);   // round-to-nearest-even
  return (u16)(r >> 16);
}

__device__ __forceinline__ v8f zero8() {
  v8f z;
#pragma unroll
  for (int i = 0; i < 8; ++i) z[i] = 0.0f;
  return z;
}

// Build a 16xbf16 fragment from two 16-byte LDS chunks (ISA 7.12.2 layouts).
__device__ __forceinline__ v16bf mk_frag(const u16* p0, const u16* p1) {
  union { us8 h[2]; v16bf v; } u;
  u.h[0] = *(const us8*)p0;
  u.h[1] = *(const us8*)p1;
  return u.v;
}

// Async global->LDS 16B copy (ISA 15.18.3 op 98, GV mode). The low 32 bits of
// a generic LDS pointer are the wave-relative LDS byte offset (ISA 10.2).
__device__ __forceinline__ void async_ld_b128(void* lds_ptr, const void* gptr) {
  unsigned loff = (unsigned)(uintptr_t)lds_ptr;
  asm volatile("global_load_async_to_lds_b128 %0, %1, off"
               :: "v"(loff), "v"(gptr) : "memory");
}
__device__ __forceinline__ void wait_async0() {
  asm volatile("s_wait_asynccnt 0x0" ::: "memory");
}

// ---------------------------------------------------------------------------
// One-shot prep: fp32 -> bf16 (vectorized, grid-stride)
// ---------------------------------------------------------------------------
__global__ __launch_bounds__(256) void cvt4(const float* __restrict__ in,
                                            u16* __restrict__ out, int n4) {
  int i = blockIdx.x * 256 + threadIdx.x;
  int stride = gridDim.x * 256;
  for (; i < n4; i += stride) {
    float4 v = ((const float4*)in)[i];
    us4 b;
    b[0] = f2bf(v.x); b[1] = f2bf(v.y); b[2] = f2bf(v.z); b[3] = f2bf(v.w);
    ((us4*)out)[i] = b;
  }
}

// ---------------------------------------------------------------------------
// One-shot prep: W (K x N, fp32) -> Wt (N x K, bf16). 32x32 LDS tiles.
// ---------------------------------------------------------------------------
__global__ __launch_bounds__(256) void transpose_cvt(const float* __restrict__ W,
                                                     u16* __restrict__ Wt,
                                                     int K, int N) {
  __shared__ float sT[32][33];
  const int tx = threadIdx.x & 31;
  const int ty = threadIdx.x >> 5;       // 0..7
  const int n0 = blockIdx.x * 32;
  const int k0 = blockIdx.y * 32;
#pragma unroll
  for (int j = 0; j < 4; ++j)
    sT[ty + j * 8][tx] = W[(size_t)(k0 + ty + j * 8) * N + n0 + tx];
  __syncthreads();
#pragma unroll
  for (int j = 0; j < 4; ++j)
    Wt[(size_t)(n0 + ty + j * 8) * K + k0 + tx] = f2bf(sT[tx][ty + j * 8]);
}

// ---------------------------------------------------------------------------
// C(MxN) = A(MxK) * Wt(NxK)^T, all bf16 in, WMMA bf16 compute, fp32 accum.
// 256 threads (8 waves), block tile 128x128, K-step 32, double-buffered LDS
// filled with async global->LDS copies (ASYNCcnt), consumed via ds_load_b128.
// Wave tile 64x32 = 4x2 subtiles of 16x16.
// ---------------------------------------------------------------------------
template <bool OUT_BF16>
__global__ __launch_bounds__(256) void gemm_wmma_bf16(
    const u16* __restrict__ A, const u16* __restrict__ Wt,
    void* __restrict__ Cout, int M, int N, int K) {
  __shared__ __align__(16) u16 sA[2][128 * 48];   // (m, k), stride 48
  __shared__ __align__(16) u16 sB[2][128 * 48];   // (n, k), stride 48

  const int t    = threadIdx.x;
  const int lane = t & 31;
  const int wv   = t >> 5;        // wave 0..7
  const int hlf  = lane >> 4;     // lane half
  const int r16  = lane & 15;
  const int wm   = wv >> 2;       // 64-row strip
  const int wn   = wv & 3;        // 32-col strip
  const int n0   = blockIdx.x * 128;
  const int m0   = blockIdx.y * 128;

  // Cooperative tile staging: 512x16B per tile per matrix, 2 per thread.
  const int grow0 = (t + 0)   >> 2, gc0 = ((t + 0)   & 3) * 8;
  const int grow1 = (t + 256) >> 2, gc1 = ((t + 256) & 3) * 8;

  auto issue_async = [&](int buf, int kt) {
    async_ld_b128(&sA[buf][grow0 * 48 + gc0], &A [(size_t)(m0 + grow0) * K + kt + gc0]);
    async_ld_b128(&sA[buf][grow1 * 48 + gc1], &A [(size_t)(m0 + grow1) * K + kt + gc1]);
    async_ld_b128(&sB[buf][grow0 * 48 + gc0], &Wt[(size_t)(n0 + grow0) * K + kt + gc0]);
    async_ld_b128(&sB[buf][grow1 * 48 + gc1], &Wt[(size_t)(n0 + grow1) * K + kt + gc1]);
  };

  v8f acc[4][2];
#pragma unroll
  for (int i = 0; i < 4; ++i)
#pragma unroll
    for (int j = 0; j < 2; ++j) acc[i][j] = zero8();

  issue_async(0, 0);
  int cur = 0;

  for (int kt = 0; kt < K; kt += 32) {
    wait_async0();                         // prev iteration's fills landed
    __syncthreads();                       // buf[cur] published to all waves
    if (kt + 32 < K) issue_async(cur ^ 1, kt + 32);   // overlap with WMMAs

    v16bf af[4], bfr[2];
#pragma unroll
    for (int ms = 0; ms < 4; ++ms) {
      const u16* arp = &sA[cur][(wm * 64 + ms * 16 + r16) * 48 + hlf * 8];
      af[ms] = mk_frag(arp, arp + 16);     // K groups {h*8..+7, +16..+23}
    }
#pragma unroll
    for (int ns = 0; ns < 2; ++ns) {
      const u16* brp = &sB[cur][(wn * 32 + ns * 16 + r16) * 48 + hlf * 8];
      bfr[ns] = mk_frag(brp, brp + 16);    // K groups {h*8..+7, +16..+23}
    }
#pragma unroll
    for (int ms = 0; ms < 4; ++ms)
#pragma unroll
      for (int ns = 0; ns < 2; ++ns)
        acc[ms][ns] = __builtin_amdgcn_wmma_f32_16x16x32_bf16(
            false, af[ms], false, bfr[ns], (short)0, acc[ms][ns], false, false);

    cur ^= 1;
  }

#pragma unroll
  for (int ms = 0; ms < 4; ++ms)
#pragma unroll
    for (int ns = 0; ns < 2; ++ns)
#pragma unroll
      for (int r = 0; r < 8; ++r) {
        int row = m0 + wm * 64 + ms * 16 + r + 8 * hlf;
        int col = n0 + wn * 32 + ns * 16 + r16;
        if (OUT_BF16)
          ((u16*)Cout)[(size_t)row * N + col] = f2bf(acc[ms][ns][r]);
        else
          ((float*)Cout)[(size_t)row * N + col] = acc[ms][ns][r];
      }
}

// ---------------------------------------------------------------------------
// Flash attention: block = (b, h, 64-query tile), 128 threads (4 waves).
// qkv is bf16 packed (B,S,3D): q @ +0, k @ +D, v @ +2D. Output bf16.
// Q/K tiles filled with async global->LDS; V transposed manually.
// ---------------------------------------------------------------------------
__global__ __launch_bounds__(128) void flash_attn_wmma(
    const u16* __restrict__ qkv, const int* __restrict__ amask,
    u16* __restrict__ attn_out) {
  __shared__ __align__(16) u16 sQ[64 * 80];        // (q_row, hd)
  __shared__ __align__(16) u16 sK[64 * 80];        // (key,   hd)
  __shared__ __align__(16) u16 sV[64 * 80];        // transposed: (hd, key)
  __shared__ __align__(16) u16 sP[4 * 16 * 80];    // per-wave P scratch

  const int t    = threadIdx.x;
  const int lane = t & 31;
  const int wv   = t >> 5;
  const int hlf  = lane >> 4;
  const int r16  = lane & 15;
  const int qt   = blockIdx.x;
  const int h    = blockIdx.y;
  const int b    = blockIdx.z;
  const int q0   = qt * 64;

  const int rowPair = t >> 1;
  const int colHalf = (t & 1) * 32;

  {
    const u16* gq = qkv + (size_t)(b * SEQ + q0 + rowPair) * (3 * D_MODEL) + h * HEAD_DIM;
#pragma unroll
    for (int j = 0; j < 4; ++j)
      async_ld_b128(&sQ[rowPair * 80 + colHalf + j * 8], gq + colHalf + j * 8);
  }
  wait_async0();
  __syncthreads();

  v16bf qa[2];
#pragma unroll
  for (int ks = 0; ks < 2; ++ks) {
    const u16* qr = &sQ[(wv * 16 + r16) * 80 + ks * 32 + hlf * 8];
    qa[ks] = mk_frag(qr, qr + 16);
  }

  float m_r[8], l_r[8];
  v8f acc_o[4];
#pragma unroll
  for (int r = 0; r < 8; ++r) { m_r[r] = NEG_INF; l_r[r] = 0.0f; }
#pragma unroll
  for (int ht = 0; ht < 4; ++ht) acc_o[ht] = zero8();

  for (int kvt = 0; kvt <= qt; ++kvt) {
    const int kv0 = kvt * 64;
    __syncthreads();                       // previous tile fully consumed
    {
      const u16* gk = qkv + (size_t)(b * SEQ + kv0 + rowPair) * (3 * D_MODEL) + D_MODEL + h * HEAD_DIM;
#pragma unroll
      for (int j = 0; j < 4; ++j)
        async_ld_b128(&sK[rowPair * 80 + colHalf + j * 8], gk + colHalf + j * 8);
      const u16* gv = qkv + (size_t)(b * SEQ + kv0 + rowPair) * (3 * D_MODEL) + 2 * D_MODEL + h * HEAD_DIM;
      u16 tmp[32];
#pragma unroll
      for (int j = 0; j < 4; ++j)
        *(uint4*)&tmp[j * 8] = *(const uint4*)&gv[colHalf + j * 8];
#pragma unroll
      for (int e = 0; e < 32; ++e)
        sV[(colHalf + e) * 80 + rowPair] = tmp[e];   // transpose V -> (hd, key)
    }
    wait_async0();
    __syncthreads();

    v8f sc[4];
#pragma unroll
    for (int nt = 0; nt < 4; ++nt) {
      sc[nt] = zero8();
#pragma unroll
      for (int ks = 0; ks < 2; ++ks) {
        const u16* kr = &sK[(nt * 16 + r16) * 80 + ks * 32 + hlf * 8];
        v16bf kb = mk_frag(kr, kr + 16);
        sc[nt] = __builtin_amdgcn_wmma_f32_16x16x32_bf16(
            false, qa[ks], false, kb, (short)0, sc[nt], false, false);
      }
    }

#pragma unroll
    for (int nt = 0; nt < 4; ++nt) {
      int kidx = kv0 + nt * 16 + r16;
      bool pad = (amask[b * SEQ + kidx] == 0);
#pragma unroll
      for (int r = 0; r < 8; ++r) {
        int qidx = q0 + wv * 16 + r + 8 * hlf;
        float svv = sc[nt][r] * 0.125f;              // 1/sqrt(64)
        sc[nt][r] = (pad || (kidx > qidx)) ? NEG_INF : svv;
      }
    }

    u16* pw = &sP[wv * 16 * 80];
    float alpha[8];
#pragma unroll
    for (int r = 0; r < 8; ++r) {
      float mx = sc[0][r];
#pragma unroll
      for (int nt = 1; nt < 4; ++nt) mx = fmaxf(mx, sc[nt][r]);
#pragma unroll
      for (int off = 8; off >= 1; off >>= 1)
        mx = fmaxf(mx, __shfl_xor(mx, off, 32));
      float mnew = fmaxf(m_r[r], mx);
      alpha[r] = __expf(m_r[r] - mnew);
      float rs = 0.0f;
#pragma unroll
      for (int nt = 0; nt < 4; ++nt) {
        float p = __expf(sc[nt][r] - mnew);
        rs += p;
        pw[(r + 8 * hlf) * 80 + nt * 16 + r16] = f2bf(p);
      }
#pragma unroll
      for (int off = 8; off >= 1; off >>= 1)
        rs += __shfl_xor(rs, off, 32);
      l_r[r] = l_r[r] * alpha[r] + rs;
      m_r[r] = mnew;
    }
#pragma unroll
    for (int ht = 0; ht < 4; ++ht)
#pragma unroll
      for (int r = 0; r < 8; ++r) acc_o[ht][r] *= alpha[r];

#pragma unroll
    for (int ks = 0; ks < 2; ++ks) {
      const u16* pr = &pw[r16 * 80 + ks * 32 + hlf * 8];
      v16bf pa = mk_frag(pr, pr + 16);
#pragma unroll
      for (int ht = 0; ht < 4; ++ht) {
        const u16* vr = &sV[(ht * 16 + r16) * 80 + ks * 32 + hlf * 8];
        v16bf vb = mk_frag(vr, vr + 16);
        acc_o[ht] = __builtin_amdgcn_wmma_f32_16x16x32_bf16(
            false, pa, false, vb, (short)0, acc_o[ht], false, false);
      }
    }
  }

#pragma unroll
  for (int ht = 0; ht < 4; ++ht)
#pragma unroll
    for (int r = 0; r < 8; ++r) {
      int qrow = q0 + wv * 16 + r + 8 * hlf;
      int col  = h * HEAD_DIM + ht * 16 + r16;
      attn_out[(size_t)(b * SEQ + qrow) * D_MODEL + col] = f2bf(acc_o[ht][r] / l_r[r]);
    }
}

extern "C" void kernel_launch(void* const* d_in, const int* in_sizes, int n_in,
                              void* d_out, int out_size, void* d_ws, size_t ws_size,
                              hipStream_t stream) {
  const float* x     = (const float*)d_in[0];
  const int*   amask = (const int*)d_in[1];
  const float* w_qkv = (const float*)d_in[2];
  const float* w_out = (const float*)d_in[3];
  float* out = (float*)d_out;

  const int M = BATCH * SEQ;                                   // 8192
  char* ws = (char*)d_ws;
  u16* qkv   = (u16*)ws;                      ws += (size_t)M * 3 * D_MODEL * 2;          // 48 MB
  u16* xb    = (u16*)ws;                      ws += (size_t)M * D_MODEL * 2;              // 16 MB
  u16* wqkvT = (u16*)ws;                      ws += (size_t)3 * D_MODEL * D_MODEL * 2;    //  6 MB
  u16* woutT = (u16*)ws;                      ws += (size_t)D_MODEL * D_MODEL * 2;        //  2 MB
  u16* attnb = (u16*)ws;                                                                  // 16 MB

  // Prep: bf16 conversion + weight transposition (one-shot, bandwidth-trivial)
  cvt4<<<1024, 256, 0, stream>>>(x, xb, M * D_MODEL / 4);
  transpose_cvt<<<dim3((3 * D_MODEL) / 32, D_MODEL / 32), 256, 0, stream>>>(
      w_qkv, wqkvT, D_MODEL, 3 * D_MODEL);
  transpose_cvt<<<dim3(D_MODEL / 32, D_MODEL / 32), 256, 0, stream>>>(
      w_out, woutT, D_MODEL, D_MODEL);

  // 1) qkv = x @ w_qkv  (bf16 out)
  gemm_wmma_bf16<true><<<dim3((3 * D_MODEL) / 128, M / 128), 256, 0, stream>>>(
      xb, wqkvT, (void*)qkv, M, 3 * D_MODEL, D_MODEL);
  // 2) flash attention (bf16 out)
  flash_attn_wmma<<<dim3(SEQ / 64, N_HEADS, BATCH), 128, 0, stream>>>(qkv, amask, attnb);
  // 3) out = attn @ w_out (fp32 out)
  gemm_wmma_bf16<false><<<dim3(D_MODEL / 128, M / 128), 256, 0, stream>>>(
      attnb, woutT, (void*)out, M, D_MODEL, D_MODEL);
}
